// kNNModel_59622736003650
// MI455X (gfx1250) — matrix operand order repaired and checked by reading, not compiled, not run on hardware
//
#include <hip/hip_runtime.h>
#include <hip/hip_bf16.h>

typedef __attribute__((ext_vector_type(16))) __bf16 v16bf;
typedef __attribute__((ext_vector_type(8)))  __bf16 v8bf;
typedef __attribute__((ext_vector_type(8)))  float  v8f;

#define KDIM      512
#define N_TEST    4096
#define N_TRAIN   100000
#define KNN       5
#define MT        64                 // test rows per workgroup
#define NPI       32                 // train rows per n-iteration
#define NW        8                  // waves per workgroup (256 threads, wave32)
#define NSLICE    8
#define SLICE_SZ  ((N_TRAIN + NSLICE - 1) / NSLICE)   // 12500
#define CAND_PER_ROW (NSLICE * 2 * 2 * KNN)           // 160: slices x col-chunk x lane-half x 5
#define BIGF      3.402823466e38f
#define EPS       1e-12f

// LDS layout (dynamic):
//   xs_hi [MT][KDIM] bf16   64 KB
//   xs_lo [MT][KDIM] bf16   64 KB
//   ts_hi [NPI][KDIM] bf16  32 KB
//   ts_lo [NPI][KDIM] bf16  32 KB
//   dsc   [NW][16][17] f32  8.5 KB  (17 stride: bank-conflict padding)
//   xsq_s [MT] f32
#define SMEM_BYTES ((size_t)(2*MT*KDIM + 2*NPI*KDIM) * 2 + (size_t)(NW*16*17 + MT) * 4)

// ---------------- row squared-norms ----------------
__global__ __launch_bounds__(256) void rownorm_kernel(const float* __restrict__ data,
                                                      float* __restrict__ out, int rows) {
  int wave = threadIdx.x >> 5, lane = threadIdx.x & 31;
  int row = blockIdx.x * 8 + wave;
  if (row >= rows) return;
  const float4* p4 = (const float4*)(data + (size_t)row * KDIM);
  float s = 0.f;
  for (int k = lane; k < KDIM / 4; k += 32) {
    float4 v = p4[k];
    s = fmaf(v.x, v.x, fmaf(v.y, v.y, fmaf(v.z, v.z, fmaf(v.w, v.w, s))));
  }
  #pragma unroll
  for (int off = 16; off; off >>= 1) s += __shfl_xor(s, off, 32);
  if (lane == 0) out[row] = s;
}

// ---------------- fused distance GEMM + per-wave top-5 ----------------
__global__ __launch_bounds__(256) void knn_dist_kernel(const float* __restrict__ x,
                                                       const float* __restrict__ train,
                                                       const float* __restrict__ tsq,
                                                       const float* __restrict__ xsq,
                                                       float* __restrict__ cand_d,
                                                       int* __restrict__ cand_i) {
  extern __shared__ char smem[];
  __bf16* xs_hi = (__bf16*)smem;
  __bf16* xs_lo = xs_hi + MT * KDIM;
  __bf16* ts_hi = xs_lo + MT * KDIM;
  __bf16* ts_lo = ts_hi + NPI * KDIM;
  float*  dsc   = (float*)(ts_lo + NPI * KDIM);
  float*  xsq_s = dsc + NW * 16 * 17;

  const int tid  = threadIdx.x;
  const int wave = tid >> 5;
  const int lane = tid & 31;
  const int rg   = wave >> 1;   // row-group 0..3 (16 test rows each)
  const int cc   = wave & 1;    // col-chunk 0..1 (16 train cols each)
  const int m0   = blockIdx.x * MT;
  const int slice = blockIdx.y;
  const int nbeg = slice * SLICE_SZ;
  const int nend = (nbeg + SLICE_SZ < N_TRAIN) ? nbeg + SLICE_SZ : N_TRAIN;

  // ---- stage x tile: f32 global -> bf16 hi/lo LDS ----
  const int K4 = KDIM / 4;
  for (int t = tid; t < MT * K4; t += 256) {
    int r  = t >> 7;          // / 128
    int k4 = t & 127;
    float4 v = ((const float4*)(x + (size_t)(m0 + r) * KDIM))[k4];
    int k = k4 * 4;
    float vv[4] = {v.x, v.y, v.z, v.w};
    #pragma unroll
    for (int q = 0; q < 4; ++q) {
      float f = vv[q];
      __bf16 h = (__bf16)f;
      xs_hi[r * KDIM + k + q] = h;
      xs_lo[r * KDIM + k + q] = (__bf16)(f - (float)h);
    }
  }
  if (tid < MT) xsq_s[tid] = xsq[m0 + tid];
  __syncthreads();

  // fragment source pointers (fixed per wave/lane)
  const __bf16* xrh = xs_hi + (size_t)(rg * 16 + (lane & 15)) * KDIM;
  const __bf16* xrl = xs_lo + (size_t)(rg * 16 + (lane & 15)) * KDIM;
  const __bf16* trh = ts_hi + (size_t)(cc * 16 + (lane & 15)) * KDIM;
  const __bf16* trl = ts_lo + (size_t)(cc * 16 + (lane & 15)) * KDIM;
  const int abase = (lane & 16) ? 8 : 0;    // A-frag K offsets: {abase..+7, abase+16..+23}
  const int bbase = (lane & 16) ? 8 : 0;    // B-frag K offsets: {bbase..+7, bbase+16..+23}

  // per-lane top-5: lane L and L+16 both own test row (L&15), split columns 0..7 / 8..15
  float bd[KNN]; int bi[KNN];
  #pragma unroll
  for (int s = 0; s < KNN; ++s) { bd[s] = BIGF; bi[s] = -1; }

  // ---- software-pipelined train staging: next chunk in registers while WMMA runs ----
  float4 stg[16];
  auto load_chunk = [&](int n0) {
    #pragma unroll
    for (int t = 0; t < 16; ++t) {
      int idx = tid + t * 256;          // over NPI*K4 = 4096 float4s
      int c = idx >> 7, k4 = idx & 127;
      int trow = n0 + c;
      stg[t] = (trow < nend) ? ((const float4*)(train + (size_t)trow * KDIM))[k4]
                             : make_float4(0.f, 0.f, 0.f, 0.f);
    }
  };
  auto store_chunk = [&]() {
    #pragma unroll
    for (int t = 0; t < 16; ++t) {
      int idx = tid + t * 256;
      int c = idx >> 7, k4 = idx & 127;
      int k = k4 * 4;
      float vv[4] = {stg[t].x, stg[t].y, stg[t].z, stg[t].w};
      #pragma unroll
      for (int q = 0; q < 4; ++q) {
        float f = vv[q];
        __bf16 h = (__bf16)f;
        ts_hi[c * KDIM + k + q] = h;
        ts_lo[c * KDIM + k + q] = (__bf16)(f - (float)h);
      }
    }
  };

  load_chunk(nbeg);

  for (int n0 = nbeg; n0 < nend; n0 += NPI) {
    store_chunk();
    __syncthreads();
    if (n0 + NPI < nend) load_chunk(n0 + NPI);   // global loads overlap the WMMA burst

    // ---- cross = x_hi*t_hi + x_hi*t_lo + x_lo*t_hi via bf16 WMMA, 2 accumulator chains ----
    v8f acc0 = {0.f, 0.f, 0.f, 0.f, 0.f, 0.f, 0.f, 0.f};
    v8f acc1 = {0.f, 0.f, 0.f, 0.f, 0.f, 0.f, 0.f, 0.f};
    #pragma unroll 4
    for (int kc = 0; kc < KDIM; kc += 64) {
      {
        v16bf Ah, Al;
        ((v8bf*)&Ah)[0] = *(const v8bf*)(xrh + kc + abase);
        ((v8bf*)&Ah)[1] = *(const v8bf*)(xrh + kc + abase + 16);
        ((v8bf*)&Al)[0] = *(const v8bf*)(xrl + kc + abase);
        ((v8bf*)&Al)[1] = *(const v8bf*)(xrl + kc + abase + 16);
        v16bf Bh, Bl;
        ((v8bf*)&Bh)[0] = *(const v8bf*)(trh + kc + bbase);
        ((v8bf*)&Bh)[1] = *(const v8bf*)(trh + kc + bbase + 16);
        ((v8bf*)&Bl)[0] = *(const v8bf*)(trl + kc + bbase);
        ((v8bf*)&Bl)[1] = *(const v8bf*)(trl + kc + bbase + 16);
        acc0 = __builtin_amdgcn_wmma_f32_16x16x32_bf16(false, Ah, false, Bh, (short)0, acc0, false, false);
        acc0 = __builtin_amdgcn_wmma_f32_16x16x32_bf16(false, Ah, false, Bl, (short)0, acc0, false, false);
        acc0 = __builtin_amdgcn_wmma_f32_16x16x32_bf16(false, Al, false, Bh, (short)0, acc0, false, false);
      }
      {
        int kd = kc + 32;
        v16bf Ah, Al;
        ((v8bf*)&Ah)[0] = *(const v8bf*)(xrh + kd + abase);
        ((v8bf*)&Ah)[1] = *(const v8bf*)(xrh + kd + abase + 16);
        ((v8bf*)&Al)[0] = *(const v8bf*)(xrl + kd + abase);
        ((v8bf*)&Al)[1] = *(const v8bf*)(xrl + kd + abase + 16);
        v16bf Bh, Bl;
        ((v8bf*)&Bh)[0] = *(const v8bf*)(trh + kd + bbase);
        ((v8bf*)&Bh)[1] = *(const v8bf*)(trh + kd + bbase + 16);
        ((v8bf*)&Bl)[0] = *(const v8bf*)(trl + kd + bbase);
        ((v8bf*)&Bl)[1] = *(const v8bf*)(trl + kd + bbase + 16);
        acc1 = __builtin_amdgcn_wmma_f32_16x16x32_bf16(false, Ah, false, Bh, (short)0, acc1, false, false);
        acc1 = __builtin_amdgcn_wmma_f32_16x16x32_bf16(false, Ah, false, Bl, (short)0, acc1, false, false);
        acc1 = __builtin_amdgcn_wmma_f32_16x16x32_bf16(false, Al, false, Bh, (short)0, acc1, false, false);
      }
    }

    // ---- distances -> padded LDS tile (same-wave, DS in-order) ----
    {
      int nj = lane & 15;
      int colIdx = n0 + cc * 16 + nj;
      bool valid = colIdx < nend;
      float tq = valid ? tsq[colIdx] : 0.f;
      int mhalf = (lane >> 4) << 3;
      #pragma unroll
      for (int v = 0; v < 8; ++v) {
        int ml = mhalf + v;                        // C layout: lanes0-15 -> M=v, lanes16-31 -> M=8+v
        float d2 = xsq_s[rg * 16 + ml] + tq - 2.0f * (acc0[v] + acc1[v]);
        d2 = fmaxf(d2, EPS);
        dsc[(wave * 16 + ml) * 17 + nj] = valid ? d2 : BIGF;
      }
    }

    // ---- top-5 insertion: all 32 lanes active, 8 columns each ----
    {
      int myrow = lane & 15;
      int jbase = (lane >> 4) * 8;
      int rowb = (wave * 16 + myrow) * 17 + jbase;
      #pragma unroll
      for (int j = 0; j < 8; ++j) {
        float d = dsc[rowb + j];
        if (d < bd[KNN - 1]) {
          bd[KNN - 1] = d;
          bi[KNN - 1] = n0 + cc * 16 + jbase + j;
          #pragma unroll
          for (int t = KNN - 1; t > 0; --t) {
            if (bd[t] < bd[t - 1]) {
              float td = bd[t]; bd[t] = bd[t - 1]; bd[t - 1] = td;
              int   ti = bi[t]; bi[t] = bi[t - 1]; bi[t - 1] = ti;
            }
          }
        }
      }
    }
    __syncthreads();   // protect ts_* overwrite next iteration
  }

  // ---- emit per-(row, slice, col-chunk, lane-half) candidates ----
  {
    int R = m0 + rg * 16 + (lane & 15);
    int sub = (slice * 2 + cc) * 2 + (lane >> 4);
    size_t base = (size_t)R * CAND_PER_ROW + (size_t)sub * KNN;
    #pragma unroll
    for (int s = 0; s < KNN; ++s) {
      cand_d[base + s] = bd[s];
      cand_i[base + s] = bi[s];
    }
  }
}

// ---------------- merge candidates + majority vote ----------------
__global__ __launch_bounds__(256) void knn_vote_kernel(const float* __restrict__ cand_d,
                                                       const int* __restrict__ cand_i,
                                                       const int* __restrict__ labels,
                                                       int* __restrict__ out) {
  int row = blockIdx.x * blockDim.x + threadIdx.x;
  if (row >= N_TEST) return;
  float cd[CAND_PER_ROW]; int ci[CAND_PER_ROW];
  for (int t = 0; t < CAND_PER_ROW; ++t) {
    cd[t] = cand_d[(size_t)row * CAND_PER_ROW + t];
    ci[t] = cand_i[(size_t)row * CAND_PER_ROW + t];
  }
  int lab[KNN];
  for (int s = 0; s < KNN; ++s) {
    float best = BIGF; int bj = 0; int bidx = 0x7fffffff;
    for (int t = 0; t < CAND_PER_ROW; ++t) {
      if (cd[t] < best || (cd[t] == best && ci[t] < bidx)) {
        best = cd[t]; bj = t; bidx = ci[t];
      }
    }
    lab[s] = (bidx >= 0 && bidx < N_TRAIN) ? labels[bidx] : 0;
    cd[bj] = BIGF;
  }
  int bestLab = 0x7fffffff, bestC = 0;
  #pragma unroll
  for (int i = 0; i < KNN; ++i) {
    int c = 0;
    #pragma unroll
    for (int j = 0; j < KNN; ++j) c += (lab[j] == lab[i]);
    if (c > bestC || (c == bestC && lab[i] < bestLab)) { bestC = c; bestLab = lab[i]; }
  }
  out[row] = bestLab;
}

// ---------------- launch ----------------
extern "C" void kernel_launch(void* const* d_in, const int* in_sizes, int n_in,
                              void* d_out, int out_size, void* d_ws, size_t ws_size,
                              hipStream_t stream) {
  const float* x      = (const float*)d_in[0];
  const float* train  = (const float*)d_in[1];
  const int*   labels = (const int*)d_in[2];
  int* out = (int*)d_out;

  // workspace layout (256B aligned sections), total ~5.7 MB
  char* ws = (char*)d_ws;
  size_t off = 0;
  auto carve = [&](size_t bytes) {
    void* p = ws + off;
    off += (bytes + 255) & ~(size_t)255;
    return p;
  };
  float* tsq    = (float*)carve((size_t)N_TRAIN * 4);
  float* xsq    = (float*)carve((size_t)N_TEST * 4);
  float* cand_d = (float*)carve((size_t)N_TEST * CAND_PER_ROW * 4);
  int*   cand_i = (int*)  carve((size_t)N_TEST * CAND_PER_ROW * 4);
  (void)ws_size; (void)in_sizes; (void)n_in; (void)out_size;

  rownorm_kernel<<<(N_TRAIN + 7) / 8, 256, 0, stream>>>(train, tsq, N_TRAIN);
  rownorm_kernel<<<(N_TEST + 7) / 8, 256, 0, stream>>>(x, xsq, N_TEST);

  (void)hipFuncSetAttribute(reinterpret_cast<const void*>(knn_dist_kernel),
                            hipFuncAttributeMaxDynamicSharedMemorySize,
                            (int)SMEM_BYTES);
  dim3 grid(N_TEST / MT, NSLICE);
  knn_dist_kernel<<<grid, 256, SMEM_BYTES, stream>>>(x, train, tsq, xsq, cand_d, cand_i);

  knn_vote_kernel<<<(N_TEST + 255) / 256, 256, 0, stream>>>(cand_d, cand_i, labels, out);
}
